// RelMultiHeadAttn_77257871720520
// MI455X (gfx1250) — matrix operand order repaired and checked by reading, not compiled, not run on hardware
//
#include <hip/hip_runtime.h>
#include <hip/hip_bf16.h>

// ---------------------------------------------------------------------------
// Transformer-XL relative multi-head attention for MI455X (gfx1250, wave32).
// All GEMMs via v_wmma_f32_16x16x32_bf16 (bf16 operands, f32 accumulate).
// Projection GEMMs register-blocked 64x32 per wave (4x2 fragments, 8 acc)
// for 2.7x better WMMA-per-load density. Attention is flash-style per
// 16-row q-tile; rel_shift handled by materializing the 17 needed BD rows
// (68KB) in WGP LDS per block.
// ---------------------------------------------------------------------------

typedef __attribute__((ext_vector_type(16))) __bf16 v16bf;
typedef __attribute__((ext_vector_type(8)))  __bf16 v8bf;
typedef __attribute__((ext_vector_type(8)))  float  v8f;

union V16 { v16bf v; __bf16 e[16]; };

#define QLEN  1024
#define KLEN  2048
#define BSZ   2
#define NHEAD 16
#define DHEAD 64
#define DM    1024

__device__ __forceinline__ __bf16 f2bf(float f) {
  unsigned u = __builtin_bit_cast(unsigned, f);
  unsigned r = u + 0x7FFFu + ((u >> 16) & 1u);   // round-to-nearest-even
  unsigned short h = (unsigned short)(r >> 16);
  return __builtin_bit_cast(__bf16, h);
}
__device__ __forceinline__ float bf2f(__bf16 b) {
  unsigned short h = __builtin_bit_cast(unsigned short, b);
  unsigned u = ((unsigned)h) << 16;
  return __builtin_bit_cast(float, u);
}

__device__ __forceinline__ v8f zero8() {
  v8f z = {0.f, 0.f, 0.f, 0.f, 0.f, 0.f, 0.f, 0.f};
  return z;
}

__device__ __forceinline__ v8f wmma_bf16(v16bf a, v16bf b, v8f c) {
  // 8 args: (neg_a, A, neg_b, B, c_mod, C, reuse_a, reuse_b)
  return __builtin_amdgcn_wmma_f32_16x16x32_bf16(false, a, false, b,
                                                 (short)0, c, false, false);
}

// A fragment (16x32 bf16). base points at (row 0, k 0) of the tile.
// Lane layout (ISA 7.12.2): row m = lane&15; K = kb + e (+16 for e>=8), kb = lane<16?0:8.
__device__ __forceinline__ v16bf load_frag_A(const __bf16* base, int lda) {
  int lane = threadIdx.x & 31;
  int m  = lane & 15;
  int kb = (lane < 16) ? 0 : 8;
  const __bf16* p = base + (size_t)m * lda + kb;
  V16 t;
  *(v8bf*)&t.e[0] = *(const v8bf*)(p);
  *(v8bf*)&t.e[8] = *(const v8bf*)(p + 16);
  return t.v;
}

// A fragment with per-lane row clamp (for the BD row i0+16 spill tile).
__device__ __forceinline__ v16bf load_frag_A_clamp(const __bf16* base, int lda,
                                                   int row0, int maxrow, int k0) {
  int lane = threadIdx.x & 31;
  int m  = lane & 15;
  int kb = (lane < 16) ? 0 : 8;
  int r = row0 + m; if (r > maxrow) r = maxrow;
  const __bf16* p = base + (size_t)r * lda + k0 + kb;
  V16 t;
  *(v8bf*)&t.e[0] = *(const v8bf*)(p);
  *(v8bf*)&t.e[8] = *(const v8bf*)(p + 16);
  return t.v;
}

// B fragment (32x16 bf16) from an [N][K]-style buffer: B[k][j] = base[j*ldb + k].
// Lane layout: col j = lane&15; K = kb + e (+16 for e>=8), kb = lane<16?0:8.
__device__ __forceinline__ v16bf load_frag_B(const __bf16* base, int ldb) {
  int lane = threadIdx.x & 31;
  int j  = lane & 15;
  int kb = (lane < 16) ? 0 : 8;
  const __bf16* p = base + (size_t)j * ldb + kb;
  V16 t;
  *(v8bf*)&t.e[0] = *(const v8bf*)(p);
  *(v8bf*)&t.e[8] = *(const v8bf*)(p + 16);
  return t.v;
}

// ---- register-blocked 64(M) x 32(N) GEMM body: 4x2 tiles, 8 accumulators --
// A points at (row0, 0) with leading dim lda; Bt points at (col0, 0) of the
// [N][K] transposed operand with leading dim ldb.
__device__ __forceinline__ void gemm_block_4x2(const __bf16* __restrict__ A, int lda,
                                               const __bf16* __restrict__ Bt, int ldb,
                                               int K, v8f acc[4][2]) {
#pragma unroll
  for (int mt = 0; mt < 4; ++mt)
#pragma unroll
    for (int nt = 0; nt < 2; ++nt) acc[mt][nt] = zero8();

  for (int k0 = 0; k0 < K; k0 += 32) {
    // prefetch a few K-steps ahead (global_prefetch_b8)
    if (k0 + 256 < K) {
      __builtin_prefetch(A + k0 + 256, 0, 0);
      __builtin_prefetch(Bt + k0 + 256, 0, 0);
    }
    v16bf af[4], bf[2];
#pragma unroll
    for (int mt = 0; mt < 4; ++mt)
      af[mt] = load_frag_A(A + (size_t)(16 * mt) * lda + k0, lda);
#pragma unroll
    for (int nt = 0; nt < 2; ++nt)
      bf[nt] = load_frag_B(Bt + (size_t)(16 * nt) * ldb + k0, ldb);
#pragma unroll
    for (int mt = 0; mt < 4; ++mt)
#pragma unroll
      for (int nt = 0; nt < 2; ++nt)
        acc[mt][nt] = wmma_bf16(af[mt], bf[nt], acc[mt][nt]);
  }
}

__device__ __forceinline__ float hmax16(float x) {
  x = fmaxf(x, __shfl_xor(x, 1, 32));
  x = fmaxf(x, __shfl_xor(x, 2, 32));
  x = fmaxf(x, __shfl_xor(x, 4, 32));
  x = fmaxf(x, __shfl_xor(x, 8, 32));
  return x;
}
__device__ __forceinline__ float hsum16(float x) {
  x += __shfl_xor(x, 1, 32);
  x += __shfl_xor(x, 2, 32);
  x += __shfl_xor(x, 4, 32);
  x += __shfl_xor(x, 8, 32);
  return x;
}

// ------------------------- conversion kernels ------------------------------

__global__ void k_cvt(const float* __restrict__ in, __bf16* __restrict__ out, int n) {
  int i = blockIdx.x * 256 + threadIdx.x;
  if (i < n) out[i] = f2bf(in[i]);
}

// out[n][k] = in[k][n]  (transpose K x N weight -> N x K bf16)
__global__ void k_cvt_t(const float* __restrict__ in, __bf16* __restrict__ out,
                        int K, int N) {
  int idx = blockIdx.x * 256 + threadIdx.x;
  if (idx < K * N) {
    int k = idx / N, n = idx - k * N;
    out[(size_t)n * K + k] = f2bf(in[idx]);
  }
}

// ------------------------- projection GEMMs --------------------------------

// q = w[-qlen:] @ Wq ; writes q + r_w_bias and q + r_r_bias as bf16.
// grid: (M/64, N/32), 1 wave per block.
__global__ __launch_bounds__(32)
void k_gemm_q(const __bf16* __restrict__ wb, const __bf16* __restrict__ wqt,
              const float* __restrict__ rwb, const float* __restrict__ rrb,
              __bf16* __restrict__ qw, __bf16* __restrict__ qr) {
  int row0 = blockIdx.x * 64, col0 = blockIdx.y * 32;
  v8f acc[4][2];
  gemm_block_4x2(wb + (size_t)(2048 + row0) * DM, DM,
                 wqt + (size_t)col0 * DM, DM, DM, acc);
  int lane = threadIdx.x & 31, nc = lane & 15, hi = lane >> 4;
#pragma unroll
  for (int mt = 0; mt < 4; ++mt)
#pragma unroll
    for (int nt = 0; nt < 2; ++nt)
#pragma unroll
      for (int v = 0; v < 8; ++v) {
        int R = row0 + 16 * mt + v + 8 * hi, C = col0 + 16 * nt + nc;
        float x = acc[mt][nt][v];
        qw[(size_t)R * DM + C] = f2bf(x + rwb[C]);
        qr[(size_t)R * DM + C] = f2bf(x + rrb[C]);
      }
}

// kv = w @ Wkv ; K -> [row][1024] bf16, V -> transposed [b][c][key] bf16.
__global__ __launch_bounds__(32)
void k_gemm_kv(const __bf16* __restrict__ wb, const __bf16* __restrict__ wkvt,
               __bf16* __restrict__ kmat, __bf16* __restrict__ vt) {
  int row0 = blockIdx.x * 64, col0 = blockIdx.y * 32;
  v8f acc[4][2];
  gemm_block_4x2(wb + (size_t)row0 * DM, DM,
                 wkvt + (size_t)col0 * DM, DM, DM, acc);
  int lane = threadIdx.x & 31, nc = lane & 15, hi = lane >> 4;
#pragma unroll
  for (int mt = 0; mt < 4; ++mt)
#pragma unroll
    for (int nt = 0; nt < 2; ++nt)
#pragma unroll
      for (int v = 0; v < 8; ++v) {
        int R = row0 + 16 * mt + v + 8 * hi, C = col0 + 16 * nt + nc;
        __bf16 h = f2bf(acc[mt][nt][v]);
        if (C < DM) {
          kmat[(size_t)R * DM + C] = h;
        } else {
          // rows of kv are (key*2 + b); store V transposed per batch: vt[b][c][key]
          vt[(size_t)(R & 1) * (DM * KLEN) + (size_t)(C - DM) * KLEN + (R >> 1)] = h;
        }
      }
}

// generic  C_bf16 = A(bf16, MxK) @ Bt(bf16, NxK)   (used for r @ Wr)
__global__ __launch_bounds__(32)
void k_gemm_b(const __bf16* __restrict__ A, const __bf16* __restrict__ Bt,
              __bf16* __restrict__ out) {
  int row0 = blockIdx.x * 64, col0 = blockIdx.y * 32;
  v8f acc[4][2];
  gemm_block_4x2(A + (size_t)row0 * DM, DM, Bt + (size_t)col0 * DM, DM, DM, acc);
  int lane = threadIdx.x & 31, nc = lane & 15, hi = lane >> 4;
#pragma unroll
  for (int mt = 0; mt < 4; ++mt)
#pragma unroll
    for (int nt = 0; nt < 2; ++nt)
#pragma unroll
      for (int v = 0; v < 8; ++v)
        out[(size_t)(row0 + 16 * mt + v + 8 * hi) * DM + (col0 + 16 * nt + nc)] =
            f2bf(acc[mt][nt][v]);
}

// final projection, fp32 output
__global__ __launch_bounds__(32)
void k_gemm_f32(const __bf16* __restrict__ A, const __bf16* __restrict__ Bt,
                float* __restrict__ out) {
  int row0 = blockIdx.x * 64, col0 = blockIdx.y * 32;
  v8f acc[4][2];
  gemm_block_4x2(A + (size_t)row0 * DM, DM, Bt + (size_t)col0 * DM, DM, DM, acc);
  int lane = threadIdx.x & 31, nc = lane & 15, hi = lane >> 4;
#pragma unroll
  for (int mt = 0; mt < 4; ++mt)
#pragma unroll
    for (int nt = 0; nt < 2; ++nt)
#pragma unroll
      for (int v = 0; v < 8; ++v)
        out[(size_t)(row0 + 16 * mt + v + 8 * hi) * DM + (col0 + 16 * nt + nc)] =
            acc[mt][nt][v];
}

// ------------------------- fused attention ---------------------------------
// grid: (QLEN/16, BSZ*NHEAD), one wave32 per block handling a 16-row q tile.
__global__ __launch_bounds__(32)
void k_attn(const __bf16* __restrict__ qw, const __bf16* __restrict__ qr,
            const __bf16* __restrict__ kmat, const __bf16* __restrict__ vt,
            const __bf16* __restrict__ rk, __bf16* __restrict__ av) {
  extern __shared__ __bf16 smem[];
  __bf16* BDl = smem;                 // [17][KLEN]  bf16 (68 KB)
  __bf16* Pl  = smem + 17 * KLEN;     // [16][32]    bf16 (P staging, A layout src)

  const int i0   = blockIdx.x * 16;
  const int b    = blockIdx.y & 1;
  const int n    = blockIdx.y >> 1;
  const int lane = threadIdx.x & 31;
  const int nc   = lane & 15;
  const int hi   = lane >> 4;
  const size_t hoff = (size_t)n * DHEAD;

  // q matrices are [qlen][bsz][1024]: query row i of batch b at b*1024 + i*2048.
  const __bf16* qrb = qr + (size_t)b * DM + hoff;

  // ---- Phase 1: BD rows i0..i0+16 (rel_shift source) into LDS via WMMA ----
  for (int t = 0; t < 2; ++t) {
    v16bf a0 = load_frag_A_clamp(qrb, 2 * DM, i0 + 16 * t, QLEN - 1, 0);
    v16bf a1 = load_frag_A_clamp(qrb, 2 * DM, i0 + 16 * t, QLEN - 1, 32);
    for (int j0 = 0; j0 < KLEN; j0 += 16) {
      const __bf16* bb = rk + (size_t)j0 * DM + hoff;  // B[k][j] = rk[j0+j][n*64+k]
      v8f acc = wmma_bf16(a1, load_frag_B(bb + 32, DM),
                wmma_bf16(a0, load_frag_B(bb + 0,  DM), zero8()));
#pragma unroll
      for (int v = 0; v < 8; ++v) {
        int lr = 16 * t + v + 8 * hi;
        if (lr <= 16) BDl[(size_t)lr * KLEN + j0 + nc] = f2bf(acc[v]);
      }
    }
  }
  __syncthreads();

  // ---- Phase 2: flash attention over key blocks of 32 ----
  const __bf16* qwb = qw + (size_t)b * DM + hoff;
  v16bf aw0 = load_frag_A_clamp(qwb, 2 * DM, i0, QLEN - 1, 0);
  v16bf aw1 = load_frag_A_clamp(qwb, 2 * DM, i0, QLEN - 1, 32);
  const __bf16* kbase = kmat + (size_t)b * DM + hoff;          // key j at + j*2048
  const __bf16* vbase = vt + (size_t)b * (DM * KLEN) + hoff * KLEN;

  float rmax[8], rsum[8];
  v8f o[4];
#pragma unroll
  for (int v = 0; v < 8; ++v) { rmax[v] = -3.0e38f; rsum[v] = 0.f; }
#pragma unroll
  for (int t = 0; t < 4; ++t) o[t] = zero8();

  const float scale = 0.125f;  // 1/sqrt(64)

  for (int kb0 = 0; kb0 < KLEN; kb0 += 32) {
    float sv[2][8];
#pragma unroll
    for (int sub = 0; sub < 2; ++sub) {
      int j0 = kb0 + 16 * sub;
      const __bf16* bb = kbase + (size_t)j0 * (2 * DM);  // B[k][j] = K[j0+j][n*64+k]
      v8f acc = wmma_bf16(aw1, load_frag_B(bb + 32, 2 * DM),
                wmma_bf16(aw0, load_frag_B(bb + 0,  2 * DM), zero8()));
#pragma unroll
      for (int v = 0; v < 8; ++v) {
        int i = i0 + v + 8 * hi;
        int j = j0 + nc;
        int u = j + QLEN - i;                  // 1..3071
        int wrap = (u >= KLEN + 1) ? 1 : 0;    // flat-reshape row wrap
        int li = (i - i0) + wrap;              // local BD row 0..16
        int sj = u - wrap * (KLEN + 1);        // src_j; 0 == zero pad
        float bd = (sj == 0) ? 0.f : bf2f(BDl[(size_t)li * KLEN + (sj - 1)]);
        sv[sub][v] = (acc[v] + bd) * scale;
      }
    }
    // online softmax: per-row max over the 32 new columns
    float mnew[8], corr[8];
#pragma unroll
    for (int v = 0; v < 8; ++v) {
      float m = hmax16(fmaxf(sv[0][v], sv[1][v]));
      mnew[v] = fmaxf(rmax[v], m);
      corr[v] = __expf(rmax[v] - mnew[v]);
      rmax[v] = mnew[v];
      rsum[v] *= corr[v];
    }
#pragma unroll
    for (int t = 0; t < 4; ++t)
#pragma unroll
      for (int v = 0; v < 8; ++v) o[t][v] *= corr[v];

    float psum[8];
#pragma unroll
    for (int v = 0; v < 8; ++v) psum[v] = 0.f;
#pragma unroll
    for (int sub = 0; sub < 2; ++sub)
#pragma unroll
      for (int v = 0; v < 8; ++v) {
        float p = __expf(sv[sub][v] - mnew[v]);
        psum[v] += p;
        Pl[(size_t)(v + 8 * hi) * 32 + 16 * sub + nc] = f2bf(p);
      }
#pragma unroll
    for (int v = 0; v < 8; ++v) rsum[v] += hsum16(psum[v]);

    __syncthreads();  // LDS ordering (single wave; DS is in-order anyway)
    v16bf pf = load_frag_A(Pl, 32);  // P as A fragment (16x32 keys)
#pragma unroll
    for (int t = 0; t < 4; ++t) {
      const __bf16* vb = vbase + (size_t)(16 * t) * KLEN + kb0;  // B[k][d]=V^T
      o[t] = wmma_bf16(pf, load_frag_B(vb, KLEN), o[t]);
    }
    __syncthreads();
  }

  // finalize: divide by row sums, write attn_vec (rows i*2+b) as bf16
#pragma unroll
  for (int t = 0; t < 4; ++t) {
#pragma unroll
    for (int v = 0; v < 8; ++v) {
      float val = o[t][v] / rsum[v];
      int R = (i0 + v + 8 * hi) * 2 + b;
      av[(size_t)R * DM + hoff + 16 * t + nc] = f2bf(val);
    }
  }
}

// ------------------------- launcher ----------------------------------------

extern "C" void kernel_launch(void* const* d_in, const int* in_sizes, int n_in,
                              void* d_out, int out_size, void* d_ws, size_t ws_size,
                              hipStream_t stream) {
  (void)in_sizes; (void)n_in; (void)out_size; (void)ws_size;
  const float* w   = (const float*)d_in[0];  // [2048, 2, 1024]
  const float* r   = (const float*)d_in[1];  // [2048, 1024]
  const float* rwb = (const float*)d_in[2];  // [16, 64] flat = 1024
  const float* rrb = (const float*)d_in[3];
  const float* Wq  = (const float*)d_in[4];  // [1024, 1024]
  const float* Wkv = (const float*)d_in[5];  // [1024, 2048]
  const float* Wr  = (const float*)d_in[6];  // [1024, 1024]
  const float* Wo  = (const float*)d_in[7];  // [1024, 1024]
  float* out = (float*)d_out;                // [1024, 2, 1024]

  __bf16* ws = (__bf16*)d_ws;
  size_t off = 0;
  __bf16* wb   = ws + off; off += (size_t)4096 * DM;   // w as bf16
  __bf16* rb   = ws + off; off += (size_t)2048 * DM;   // r as bf16
  __bf16* wqt  = ws + off; off += (size_t)DM * DM;     // Wq^T
  __bf16* wkvt = ws + off; off += (size_t)2048 * DM;   // Wkv^T
  __bf16* wrt  = ws + off; off += (size_t)DM * DM;     // Wr^T
  __bf16* wot  = ws + off; off += (size_t)DM * DM;     // Wo^T
  __bf16* qw   = ws + off; off += (size_t)2048 * DM;   // q + r_w_bias
  __bf16* qr   = ws + off; off += (size_t)2048 * DM;   // q + r_r_bias
  __bf16* kmat = ws + off; off += (size_t)4096 * DM;   // K heads
  __bf16* vt   = ws + off; off += (size_t)2 * DM * KLEN; // V transposed per batch
  __bf16* rk   = ws + off; off += (size_t)2048 * DM;   // r @ Wr
  __bf16* av   = ws + off; off += (size_t)2048 * DM;   // attention output

  // fp32 -> bf16 (plain / transposed)
  k_cvt  <<<(4194304 + 255) / 256, 256, 0, stream>>>(w, wb, 4194304);
  k_cvt  <<<(2097152 + 255) / 256, 256, 0, stream>>>(r, rb, 2097152);
  k_cvt_t<<<4096, 256, 0, stream>>>(Wq,  wqt,  DM, DM);
  k_cvt_t<<<8192, 256, 0, stream>>>(Wkv, wkvt, DM, 2 * DM);
  k_cvt_t<<<4096, 256, 0, stream>>>(Wr,  wrt,  DM, DM);
  k_cvt_t<<<4096, 256, 0, stream>>>(Wo,  wot,  DM, DM);

  // projections (WMMA, 64x32 per wave)
  k_gemm_q <<<dim3(32, 32), 32, 0, stream>>>(wb, wqt, rwb, rrb, qw, qr);
  k_gemm_kv<<<dim3(64, 64), 32, 0, stream>>>(wb, wkvt, kmat, vt);
  k_gemm_b <<<dim3(32, 32), 32, 0, stream>>>(rb, wrt, rk);

  // fused rel-shift flash attention (WMMA + LDS)
  size_t lds_bytes = (size_t)(17 * KLEN + 16 * 32) * sizeof(__bf16);  // ~70.7 KB
  k_attn<<<dim3(QLEN / 16, BSZ * NHEAD), 32, lds_bytes, stream>>>(
      qw, qr, kmat, vt, rk, av);

  // output projection (WMMA, fp32 out)
  k_gemm_f32<<<dim3(32, 32), 32, 0, stream>>>(av, wot, out);
}
